// MPGAT_1400159338964
// MI455X (gfx1250) — matrix operand order repaired and checked by reading, not compiled, not run on hardware
//
#include <hip/hip_runtime.h>

// ---------------------------------------------------------------------------
// MPGAT on MI455X (gfx1250): bf16 WMMA (f32 accumulate), fp32 softmax/LN/GELU.
// wave32, v_wmma_f32_16x16x32_bf16 tiles; all B-operands pre-transposed so
// every matrix fragment is loaded as two contiguous 128-bit reads.
// ---------------------------------------------------------------------------

typedef __bf16 bf16;
typedef __attribute__((ext_vector_type(16))) __bf16 v16bf;
typedef __attribute__((ext_vector_type(8)))  float  v8f;

#define BB 2
#define WW 8
#define PP 1024
#define DES 96
#define FEAT 128
#define OBJ 64
#define BWN 16           // B*W
#define MROWS (BWN*PP)   // 16384
#define PE_ROWS 2048

union Frag32B { v16bf v; uint4 q[2]; };

static __device__ __forceinline__ v8f wmma_bf(v16bf a, v16bf b, v8f c) {
  // D(f32 16x16) = A(bf16 16x32) * B(bf16 32x16) + C   -> v_wmma_f32_16x16x32_bf16
  return __builtin_amdgcn_wmma_f32_16x16x32_bf16(false, a, false, b, (short)0, c, false, false);
}

// A-fragment (16x32 bf16, M x K), source row-major [*, ldm] (ldm multiple of 8).
// lane (g=lane/16, m=lane%16); comp j<8 -> k=32kc+8g+j, j>=8 -> k=32kc+16+8g+(j-8).
static __device__ __forceinline__ v16bf frag_A_rm(const bf16* base, int ldm, int kc, int lane) {
  int g = (lane >> 4) & 1, m = lane & 15;
  const bf16* r = base + (size_t)m * ldm + 32 * kc + 8 * g;
  Frag32B u;
  u.q[0] = *(const uint4*)(r);
  u.q[1] = *(const uint4*)(r + 16);
  return u.v;
}

// B-fragment (32x16 bf16, K x N) where source stores one row per N index
// (i.e. N x K, we multiply by its transpose): B(k,n) = baseN[n][k].
// Per lane: comp j<8 -> k = 32kc + 8g + j, j>=8 -> k = 32kc + 16 + 8g + (j-8).
static __device__ __forceinline__ v16bf frag_B_nrows(const bf16* baseN, int ldn, int kc, int lane) {
  int g = (lane >> 4) & 1, n = lane & 15;
  const bf16* r = baseN + (size_t)n * ldn + 32 * kc + 8 * g;
  Frag32B u;
  u.q[0] = *(const uint4*)(r);
  u.q[1] = *(const uint4*)(r + 16);
  return u.v;
}

// ---------------------------------------------------------------------------
__global__ __launch_bounds__(256) void k_f32_to_bf16(const float* __restrict__ in,
                                                     bf16* __restrict__ out, int n) {
  int i = blockIdx.x * 256 + threadIdx.x;
  if (i < n) out[i] = (bf16)in[i];
}

// WT[n*K + k] = (bf16)W[k*N + n]   (weights are tiny; done once)
__global__ __launch_bounds__(256) void k_transpose_w(const float* __restrict__ W,
                                                     bf16* __restrict__ WT, int K, int N) {
  int i = blockIdx.x * 256 + threadIdx.x;
  if (i < K * N) {
    int k = i / N, n = i - k * N;
    WT[(size_t)n * K + k] = (bf16)W[i];
  }
}

// pe[j][f], j in [0,2048): pos = j - 1025; f<64 sin, f>=64 cos.
__global__ __launch_bounds__(256) void k_pe(bf16* __restrict__ pe) {
  int i = blockIdx.x * 256 + threadIdx.x;           // 2048*128
  int j = i >> 7, f = i & 127;
  float inv = __expf(-(float)(f & 63) * (__logf(10000.0f) / 63.0f));
  float a = (float)(j - 1025) * inv;
  float v = (f < 64) ? __sinf(a) : __cosf(a);
  pe[i] = (bf16)v;
}

// x = inputs_bf16[16384,96] @ dim_w[96,128] + dim_b  -> x_f32, x_bf16
__global__ __launch_bounds__(128) void k_gemm_embed(const bf16* __restrict__ Xin,
                                                    const bf16* __restrict__ WT,  // [128][96]
                                                    const float* __restrict__ bias,
                                                    float* __restrict__ Xout,
                                                    bf16* __restrict__ Xbf) {
  int wave = threadIdx.x >> 5, lane = threadIdx.x & 31;
  int m0 = blockIdx.x * 16;
  const bf16* A = Xin + (size_t)m0 * DES;
  int g = lane >> 4, n = lane & 15;
  for (int nt = wave; nt < 8; nt += 4) {
    int n0 = nt * 16;
    v8f acc = {};
#pragma unroll
    for (int c = 0; c < 3; ++c)
      acc = wmma_bf(frag_A_rm(A, DES, c, lane),
                    frag_B_nrows(WT + (size_t)n0 * DES, DES, c, lane), acc);
#pragma unroll
    for (int r = 0; r < 8; ++r) {
      float v = acc[r] + bias[n0 + n];
      size_t idx = (size_t)(m0 + r + 8 * g) * FEAT + n0 + n;
      Xout[idx] = v;
      Xbf[idx] = (bf16)v;
    }
  }
}

// qkv = x_bf @ qkv_w[layer]; epilogue splits into qu=q+u_p, qv=q+v_p, k, v (bf16)
__global__ __launch_bounds__(128) void k_gemm_qkv(const bf16* __restrict__ Xb,
                                                  const bf16* __restrict__ WT,  // [384][128]
                                                  const float* __restrict__ up,
                                                  const float* __restrict__ vp,
                                                  bf16* __restrict__ QU, bf16* __restrict__ QV,
                                                  bf16* __restrict__ KB, bf16* __restrict__ VB) {
  int wave = threadIdx.x >> 5, lane = threadIdx.x & 31;
  int m0 = blockIdx.x * 16;
  int n0 = (blockIdx.y * 4 + wave) * 16;            // 24 n-tiles
  const bf16* A = Xb + (size_t)m0 * FEAT;
  v8f acc = {};
#pragma unroll
  for (int c = 0; c < 4; ++c)
    acc = wmma_bf(frag_A_rm(A, FEAT, c, lane),
                  frag_B_nrows(WT + (size_t)n0 * FEAT, FEAT, c, lane), acc);
  int g = lane >> 4, n = lane & 15;
  int nabs = n0 + n;
#pragma unroll
  for (int r = 0; r < 8; ++r) {
    int m = m0 + r + 8 * g;
    float v = acc[r];
    if (nabs < FEAT) {
      size_t idx = (size_t)m * FEAT + nabs;
      QU[idx] = (bf16)(v + up[nabs]);
      QV[idx] = (bf16)(v + vp[nabs]);
    } else if (nabs < 2 * FEAT) {
      KB[(size_t)m * FEAT + (nabs - FEAT)] = (bf16)v;
    } else {
      VB[(size_t)m * FEAT + (nabs - 2 * FEAT)] = (bf16)v;
    }
  }
}

// C[bw][1024,2048] = A[bw][1024,128] @ pe^T   (Bfull = qv@pe^T / Efull = k@pe^T)
// blockIdx.y = bw within the batch.
__global__ __launch_bounds__(128) void k_gemm_rel(const bf16* __restrict__ Abase,
                                                  const bf16* __restrict__ PE,
                                                  float* __restrict__ Cbase) {
  int t = threadIdx.x, wave = t >> 5, lane = t & 31;
  int id = blockIdx.x * 4 + wave;       // 8192 tiles (64 m x 128 n)
  int mt = id >> 7, nt = id & 127;
  const bf16* A = Abase + (size_t)blockIdx.y * PP * FEAT + (size_t)(mt * 16) * FEAT;
  const bf16* Bn = PE + (size_t)(nt * 16) * FEAT;
  v8f acc = {};
#pragma unroll
  for (int c = 0; c < 4; ++c)
    acc = wmma_bf(frag_A_rm(A, FEAT, c, lane), frag_B_nrows(Bn, FEAT, c, lane), acc);
  int g = lane >> 4, n = lane & 15;
  float* Cp = Cbase + (size_t)blockIdx.y * PP * PE_ROWS + (size_t)(mt * 16) * PE_ROWS + nt * 16;
#pragma unroll
  for (int r = 0; r < 8; ++r) Cp[(size_t)(r + 8 * g) * PE_ROWS + n] = acc[r];
}

// Fused flash-attention: AC via WMMA (K tile staged in LDS, shared by 4 waves),
// BD gathered coalesced from Bfull, Es staged coalesced into LDS (EL band),
// single online-softmax update per 32-wide k-step, P@V via WMMA with V staged
// transposed in LDS. blockIdx.y = bw within batch.
__global__ __launch_bounds__(128) void k_attn(const bf16* __restrict__ QUb,
                                              const bf16* __restrict__ KBb,
                                              const bf16* __restrict__ VBb,
                                              const float* __restrict__ Bfb,
                                              const float* __restrict__ Efb,
                                              const unsigned char* __restrict__ adjb,
                                              float* __restrict__ Ob) {
  __shared__ __align__(16) bf16 KL[32 * 136];      // K tile row-major [k][feat], padded
  __shared__ __align__(16) bf16 VT[128 * 40];      // V tile transposed: [feat][k], padded
  __shared__ float EL[32 * 65];                    // Es parallelogram [k][dq], padded
  __shared__ __align__(16) bf16 Pb[4][16 * 32];    // per-wave P tile (A-layout source)
  int t = threadIdx.x, wave = t >> 5, lane = t & 31;
  size_t bw = blockIdx.y;
  const bf16* QU = QUb + bw * PP * FEAT;
  const bf16* KB = KBb + bw * PP * FEAT;
  const bf16* VB = VBb + bw * PP * FEAT;
  const float* Bfull = Bfb + bw * PP * PE_ROWS;
  const float* Efull = Efb + bw * PP * PE_ROWS;
  const unsigned char* adj = adjb + bw * PP * PP;
  float* O = Ob + bw * PP * FEAT;

  int q0b = blockIdx.x * 64;
  int q0 = q0b + wave * 16;
  int g = lane >> 4, n = lane & 15;

  v16bf aq[4];
#pragma unroll
  for (int c = 0; c < 4; ++c) aq[c] = frag_A_rm(QU + (size_t)q0 * FEAT, FEAT, c, lane);

  float mi[8], li[8];
  v8f oacc[8];
  v8f zero = {};
#pragma unroll
  for (int r = 0; r < 8; ++r) { mi[r] = -1e30f; li[r] = 0.0f; }
#pragma unroll
  for (int c2 = 0; c2 < 8; ++c2) oacc[c2] = zero;

  const float inv_scale = 0.08838834764831845f;    // 1/sqrt(128)

  for (int k0 = 0; k0 < PP; k0 += 32) {
    __syncthreads();
    // stage K (row-major) + V (transposed), coalesced b128-friendly reads
    for (int i = t; i < 32 * 128; i += 128) {
      int kd = i >> 7, col = i & 127;
      KL[kd * 136 + col] = KB[(size_t)(k0 + kd) * FEAT + col];
      VT[col * 40 + kd]  = VB[(size_t)(k0 + kd) * FEAT + col];
    }
    // stage Es parallelogram: EL[kd][dq] = Efull[k0+kd][PP + q0b + dq - (k0+kd)]
    for (int i = t; i < 32 * 64; i += 128) {
      int kd = i >> 6, dq = i & 63;
      EL[kd * 65 + dq] = Efull[(size_t)(k0 + kd) * PE_ROWS + (PP + q0b - (k0 + kd)) + dq];
    }
    // prefetch next k-step K/V tiles (global_prefetch_b8)
    if (k0 + 32 < PP) {
      int kd2 = t >> 2, c2_ = (t & 3) * 32;
      __builtin_prefetch(&KB[(size_t)(k0 + 32 + kd2) * FEAT + c2_], 0, 0);
      __builtin_prefetch(&VB[(size_t)(k0 + 32 + kd2) * FEAT + c2_], 0, 0);
    }
    __syncthreads();

    // scores for both 16-col halves
    v8f sc0 = zero, sc1 = zero;
#pragma unroll
    for (int c = 0; c < 4; ++c) sc0 = wmma_bf(aq[c], frag_B_nrows(KL, 136, c, lane), sc0);
#pragma unroll
    for (int c = 0; c < 4; ++c) sc1 = wmma_bf(aq[c], frag_B_nrows(KL + 16 * 136, 136, c, lane), sc1);

    float sv[2][8];
#pragma unroll
    for (int h = 0; h < 2; ++h) {
      int kcol = k0 + 16 * h + n;
#pragma unroll
      for (int r = 0; r < 8; ++r) {
        int qrow = q0 + r + 8 * g;
        float bd = Bfull[(size_t)qrow * PE_ROWS + (PP + kcol - qrow)];
        float es = EL[(16 * h + n) * 65 + (qrow - q0b)];
        float s_ = (h ? sc1[r] : sc0[r]);
        float val = (s_ + bd + es) * inv_scale;
        sv[h][r] = adj[(size_t)qrow * PP + kcol] ? val : -1e9f;
      }
    }

    // one joint online-softmax update for the full 32-col step
#pragma unroll
    for (int r = 0; r < 8; ++r) {
      float a0 = sv[0][r], a1 = sv[1][r];
      float mx = fmaxf(a0, a1);
#pragma unroll
      for (int o_ = 8; o_; o_ >>= 1) mx = fmaxf(mx, __shfl_xor(mx, o_, 16));
      float nm = fmaxf(mi[r], mx);
      float alpha = __expf(mi[r] - nm);
      float p0 = __expf(a0 - nm);
      float p1 = __expf(a1 - nm);
      float ps = p0 + p1;
#pragma unroll
      for (int o_ = 8; o_; o_ >>= 1) ps += __shfl_xor(ps, o_, 16);
      li[r] = li[r] * alpha + ps;
      mi[r] = nm;
#pragma unroll
      for (int c2 = 0; c2 < 8; ++c2) oacc[c2][r] *= alpha;
      Pb[wave][(r + 8 * g) * 32 + n] = (bf16)p0;
      Pb[wave][(r + 8 * g) * 32 + 16 + n] = (bf16)p1;
    }
    __syncthreads();   // order LDS P writes before fragment reads

    v16bf pa = frag_A_rm(&Pb[wave][0], 32, 0, lane);
#pragma unroll
    for (int c2 = 0; c2 < 8; ++c2)
      oacc[c2] = wmma_bf(pa, frag_B_nrows(&VT[(size_t)(16 * c2) * 40], 40, 0, lane), oacc[c2]);
  }

#pragma unroll
  for (int c2 = 0; c2 < 8; ++c2)
#pragma unroll
    for (int r = 0; r < 8; ++r)
      O[(size_t)(q0 + r + 8 * g) * FEAT + 16 * c2 + n] = oacc[c2][r] / li[r];
}

// s = LayerNorm(a + b) * w + bias  -> f32 + bf16.  One wave per 128-dim row.
__global__ __launch_bounds__(128) void k_add_ln(const float* __restrict__ Xa,
                                                const float* __restrict__ Xb_,
                                                const float* __restrict__ w,
                                                const float* __restrict__ b,
                                                float* __restrict__ Sf,
                                                bf16* __restrict__ Sbf) {
  int wave = threadIdx.x >> 5, lane = threadIdx.x & 31;
  size_t row = (size_t)blockIdx.x * 4 + wave;
  const float* pa = Xa + row * FEAT;
  const float* pb = Xb_ + row * FEAT;
  float v[4], s = 0.0f;
#pragma unroll
  for (int i = 0; i < 4; ++i) { v[i] = pa[lane * 4 + i] + pb[lane * 4 + i]; s += v[i]; }
#pragma unroll
  for (int o_ = 16; o_; o_ >>= 1) s += __shfl_xor(s, o_, 32);
  float mean = s * (1.0f / 128.0f);
  float var = 0.0f;
#pragma unroll
  for (int i = 0; i < 4; ++i) { float d = v[i] - mean; var += d * d; }
#pragma unroll
  for (int o_ = 16; o_; o_ >>= 1) var += __shfl_xor(var, o_, 32);
  float rstd = __frsqrt_rn(var * (1.0f / 128.0f) + 1e-5f);
#pragma unroll
  for (int i = 0; i < 4; ++i) {
    int f = lane * 4 + i;
    float o = (v[i] - mean) * rstd * w[f] + b[f];
    Sf[row * FEAT + f] = o;
    Sbf[row * FEAT + f] = (bf16)o;
  }
}

// FFN GEMM 128x128; mode 1: GELU(exact erf) -> bf16 out; mode 0: +bias -> f32 out.
__global__ __launch_bounds__(256) void k_ffn(const bf16* __restrict__ A_,
                                             const bf16* __restrict__ WT,  // [128][128]
                                             const float* __restrict__ bias,
                                             float* __restrict__ Of, bf16* __restrict__ Obf,
                                             int gelu_mode) {
  int wave = threadIdx.x >> 5, lane = threadIdx.x & 31;
  int m0 = blockIdx.x * 16, n0 = wave * 16;
  const bf16* A = A_ + (size_t)m0 * FEAT;
  v8f acc = {};
#pragma unroll
  for (int c = 0; c < 4; ++c)
    acc = wmma_bf(frag_A_rm(A, FEAT, c, lane),
                  frag_B_nrows(WT + (size_t)n0 * FEAT, FEAT, c, lane), acc);
  int g = lane >> 4, n = lane & 15;
#pragma unroll
  for (int r = 0; r < 8; ++r) {
    float v = acc[r] + bias[n0 + n];
    size_t idx = (size_t)(m0 + r + 8 * g) * FEAT + n0 + n;
    if (gelu_mode) {
      v = 0.5f * v * (1.0f + erff(v * 0.70710678118654752f));
      Obf[idx] = (bf16)v;
    } else {
      Of[idx] = v;
    }
  }
}

// y[b,l,f] = max over w of x[b,w,l,f]  -> bf16
__global__ __launch_bounds__(256) void k_maxw(const float* __restrict__ X, bf16* __restrict__ Y) {
  int i = blockIdx.x * 256 + threadIdx.x;   // 2*1024*128
  int f = i & 127, l = (i >> 7) & 1023, b = i >> 17;
  float m = -3e38f;
#pragma unroll
  for (int w = 0; w < WW; ++w)
    m = fmaxf(m, X[((size_t)((b * WW + w) * PP + l)) * FEAT + f]);
  Y[i] = (bf16)m;
}

// out[2048,64] = y_bf @ out_w + out_b  -> f32
__global__ __launch_bounds__(128) void k_out(const bf16* __restrict__ Yb,
                                             const bf16* __restrict__ WT,  // [64][128]
                                             const float* __restrict__ bias,
                                             float* __restrict__ Out) {
  int wave = threadIdx.x >> 5, lane = threadIdx.x & 31;
  int id = blockIdx.x * 4 + wave;   // 512 tiles: 128 m-tiles x 4 n-tiles
  int mt = id >> 2, nt = id & 3;
  const bf16* A = Yb + (size_t)(mt * 16) * FEAT;
  v8f acc = {};
#pragma unroll
  for (int c = 0; c < 4; ++c)
    acc = wmma_bf(frag_A_rm(A, FEAT, c, lane),
                  frag_B_nrows(WT + (size_t)(nt * 16) * FEAT, FEAT, c, lane), acc);
  int g = lane >> 4, n = lane & 15;
#pragma unroll
  for (int r = 0; r < 8; ++r)
    Out[(size_t)(mt * 16 + r + 8 * g) * OBJ + nt * 16 + n] = acc[r] + bias[nt * 16 + n];
}

// ---------------------------------------------------------------------------
extern "C" void kernel_launch(void* const* d_in, const int* in_sizes, int n_in,
                              void* d_out, int out_size, void* d_ws, size_t ws_size,
                              hipStream_t stream) {
  (void)in_sizes; (void)n_in; (void)out_size;
  const float* in_x   = (const float*)d_in[0];
  const unsigned char* adj = (const unsigned char*)d_in[1];   // jnp bool_ = 1 byte
  const float* dim_w  = (const float*)d_in[2];
  const float* dim_b  = (const float*)d_in[3];
  const float* qkv_w  = (const float*)d_in[4];
  const float* u_p    = (const float*)d_in[5];
  const float* v_p    = (const float*)d_in[6];
  const float* ln1_w  = (const float*)d_in[7];
  const float* ln1_b  = (const float*)d_in[8];
  const float* ln2_w  = (const float*)d_in[9];
  const float* ln2_b  = (const float*)d_in[10];
  const float* ffn_w1 = (const float*)d_in[11];
  const float* ffn_b1 = (const float*)d_in[12];
  const float* ffn_w2 = (const float*)d_in[13];
  const float* ffn_b2 = (const float*)d_in[14];
  const float* out_w  = (const float*)d_in[15];
  const float* out_b  = (const float*)d_in[16];
  float* outp = (float*)d_out;

  char* ws = (char*)d_ws;
  size_t off = 0;
  auto alloc = [&](size_t bytes) -> void* {
    void* p = ws + off;
    off = (off + bytes + 255) & ~((size_t)255);
    return p;
  };
  bf16*  pe     = (bf16*)alloc((size_t)PE_ROWS * FEAT * 2);
  bf16*  xin_bf = (bf16*)alloc((size_t)MROWS * DES * 2);
  float* x_f    = (float*)alloc((size_t)MROWS * FEAT * 4);
  bf16*  x_bf   = (bf16*)alloc((size_t)MROWS * FEAT * 2);
  bf16*  qu     = (bf16*)alloc((size_t)MROWS * FEAT * 2);
  bf16*  qv     = (bf16*)alloc((size_t)MROWS * FEAT * 2);
  bf16*  kb     = (bf16*)alloc((size_t)MROWS * FEAT * 2);
  bf16*  vb     = (bf16*)alloc((size_t)MROWS * FEAT * 2);
  float* o_f    = (float*)alloc((size_t)MROWS * FEAT * 4);
  float* s_f    = (float*)alloc((size_t)MROWS * FEAT * 4);
  bf16*  s_bf   = (bf16*)alloc((size_t)MROWS * FEAT * 2);
  bf16*  f1_bf  = (bf16*)alloc((size_t)MROWS * FEAT * 2);
  float* f2_f   = (float*)alloc((size_t)MROWS * FEAT * 4);
  bf16*  y_bf   = (bf16*)alloc((size_t)BB * PP * FEAT * 2);
  // transposed bf16 weights
  bf16*  dimwT  = (bf16*)alloc((size_t)FEAT * DES * 2);            // [128][96]
  bf16*  qkvT   = (bf16*)alloc((size_t)2 * 3 * FEAT * FEAT * 2);   // per layer [384][128]
  bf16*  w1T    = (bf16*)alloc((size_t)2 * FEAT * FEAT * 2);
  bf16*  w2T    = (bf16*)alloc((size_t)2 * FEAT * FEAT * 2);
  bf16*  outT   = (bf16*)alloc((size_t)OBJ * FEAT * 2);            // [64][128]

  // Carve remaining ws into G pairs of (Bfull,Efull) score side-tables
  // (16 MB/pair); batch the per-(b,w) attention phase G sequences at a time.
  const size_t pair_bytes = (size_t)PP * PE_ROWS * 4 * 2;
  size_t rem = (ws_size > off) ? (ws_size - off) : 0;
  int G = (int)(rem / pair_bytes);
  if (G < 1) G = 1;          // assume harness sized ws for at least the base plan
  if (G > BWN) G = BWN;
  float* Bfull = (float*)alloc((size_t)G * PP * PE_ROWS * 4);
  float* Efull = (float*)alloc((size_t)G * PP * PE_ROWS * 4);

  // prep: bf16 inputs, pe table, transposed weights
  {
    int n = MROWS * DES;
    k_f32_to_bf16<<<(n + 255) / 256, 256, 0, stream>>>(in_x, xin_bf, n);
  }
  k_pe<<<(PE_ROWS * FEAT) / 256, 256, 0, stream>>>(pe);
  k_transpose_w<<<(DES * FEAT + 255) / 256, 256, 0, stream>>>(dim_w, dimwT, DES, FEAT);
  for (int l = 0; l < 2; ++l) {
    k_transpose_w<<<(FEAT * 3 * FEAT + 255) / 256, 256, 0, stream>>>(
        qkv_w + (size_t)l * FEAT * 3 * FEAT, qkvT + (size_t)l * 3 * FEAT * FEAT, FEAT, 3 * FEAT);
    k_transpose_w<<<(FEAT * FEAT + 255) / 256, 256, 0, stream>>>(
        ffn_w1 + (size_t)l * FEAT * FEAT, w1T + (size_t)l * FEAT * FEAT, FEAT, FEAT);
    k_transpose_w<<<(FEAT * FEAT + 255) / 256, 256, 0, stream>>>(
        ffn_w2 + (size_t)l * FEAT * FEAT, w2T + (size_t)l * FEAT * FEAT, FEAT, FEAT);
  }
  k_transpose_w<<<(FEAT * OBJ + 255) / 256, 256, 0, stream>>>(out_w, outT, FEAT, OBJ);

  k_gemm_embed<<<MROWS / 16, 128, 0, stream>>>(xin_bf, dimwT, dim_b, x_f, x_bf);

  for (int layer = 0; layer < 2; ++layer) {
    k_gemm_qkv<<<dim3(MROWS / 16, 6), 128, 0, stream>>>(
        x_bf, qkvT + (size_t)layer * 3 * FEAT * FEAT,
        u_p + layer * FEAT, v_p + layer * FEAT, qu, qv, kb, vb);

    for (int bw0 = 0; bw0 < BWN; bw0 += G) {
      int gc = (bw0 + G <= BWN) ? G : (BWN - bw0);
      size_t mo = (size_t)bw0 * PP * FEAT;
      k_gemm_rel<<<dim3(2048, gc), 128, 0, stream>>>(qv + mo, pe, Bfull);
      k_gemm_rel<<<dim3(2048, gc), 128, 0, stream>>>(kb + mo, pe, Efull);
      k_attn<<<dim3(PP / 64, gc), 128, 0, stream>>>(
          qu + mo, kb + mo, vb + mo, Bfull, Efull,
          adj + (size_t)bw0 * PP * PP, o_f + mo);
    }
    k_add_ln<<<MROWS / 4, 128, 0, stream>>>(o_f, x_f, ln1_w + layer * FEAT,
                                            ln1_b + layer * FEAT, s_f, s_bf);
    k_ffn<<<MROWS / 16, 256, 0, stream>>>(s_bf, w1T + (size_t)layer * FEAT * FEAT,
                                          ffn_b1 + layer * FEAT, nullptr, f1_bf, 1);
    k_ffn<<<MROWS / 16, 256, 0, stream>>>(f1_bf, w2T + (size_t)layer * FEAT * FEAT,
                                          ffn_b2 + layer * FEAT, f2_f, nullptr, 0);
    k_add_ln<<<MROWS / 4, 128, 0, stream>>>(f2_f, s_f, ln2_w + layer * FEAT,
                                            ln2_b + layer * FEAT, x_f, x_bf);
  }

  k_maxw<<<(BB * PP * FEAT) / 256, 256, 0, stream>>>(x_f, y_bf);
  k_out<<<128, 128, 0, stream>>>(y_bf, outT, out_b, outp);
}
